// NonAttentiveTacatron_59450937312002
// MI455X (gfx1250) — matrix operand order repaired and checked by reading, not compiled, not run on hardware
//
// NonAttentiveTacotron forward for MI455X (gfx1250), wave32 + WMMA f16.
// GEMM-shaped work (convs-as-tap-GEMMs, LSTM projections, attention einsum,
// postnet) runs through v_wmma_f32_16x16x32_f16 with LDS-staged fragments.
// Tile staging: global_load_async_to_lds_b128 (ASYNCcnt) + global_prefetch.
// All GEMM operands are padded so lda/ldw/K are multiples of 8 -> no scalar
// fallback path exists in the binary.
#include <hip/hip_runtime.h>
#include <hip/hip_bf16.h>
#include <math.h>

typedef __attribute__((ext_vector_type(16))) _Float16 v16h;
typedef __attribute__((ext_vector_type(8)))  _Float16 v8h;
typedef __attribute__((ext_vector_type(8)))  float    v8f;

#define B_    32
#define S_    256
#define T_    1018
#define EPH   512
#define ESPK  64
#define FD    576      // 2*ENCH + ESPK
#define PD    32
#define MD    608      // FD + PD
#define NMEL  80
#define ENCH  256
#define DH    1024
#define PN    256
#define PC    512
#define RNGK  584      // FD + 1 (=577) padded up to a multiple of 8
#define NTS   (B_*S_)  // 8192 tokens
#define NTT   (B_*T_)  // 32576 frames
#define LDT   40       // LDS tile row stride in halves (80B, keeps 16B align)

#define ACT_NONE 0
#define ACT_RELU 1
#define ACT_TANH 2

__device__ __forceinline__ float sigm(float x) { return 1.f / (1.f + __expf(-x)); }

// --- CDNA5 async global->LDS copy (16B per lane), tracked by ASYNCcnt -------
__device__ __forceinline__ void async_g2l_b128(const _Float16* g, _Float16* l) {
#if defined(__gfx1250__)
  asm volatile("global_load_async_to_lds_b128 %0, %1, off"
               :: "v"((unsigned)(unsigned long long)l),
                  "v"((unsigned long long)g)
               : "memory");
#else
  *(v8h*)l = *(const v8h*)g;
#endif
}
__device__ __forceinline__ void async_wait_all() {
#if defined(__gfx1250__)
  asm volatile("s_wait_asynccnt 0x0" ::: "memory");
#endif
}

// ---------------------------------------------------------------- small utils
__global__ void k_cvt16(const float* __restrict__ src, _Float16* __restrict__ dst,
                        int n, int cols, int rs, int cs, int dld) {
  int i = blockIdx.x * blockDim.x + threadIdx.x;
  if (i >= n) return;
  int r = i / cols, c = i % cols;
  dst[(long)r * dld + c] = (_Float16)src[(long)r * rs + (long)c * cs];
}
__global__ void k_addv(const float* a, const float* b, float* o, int n) {
  int i = blockIdx.x * blockDim.x + threadIdx.x;
  if (i < n) o[i] = a[i] + b[i];
}
__global__ void k_fillf(float* p, float v, int n) {
  int i = blockIdx.x * blockDim.x + threadIdx.x;
  if (i < n) p[i] = v;
}
__global__ void k_fillh(_Float16* p, int n) {
  int i = blockIdx.x * blockDim.x + threadIdx.x;
  if (i < n) p[i] = (_Float16)0.f;
}
__global__ void k_copy16(const _Float16* __restrict__ src, int ls,
                         _Float16* __restrict__ dst, int ld, int cols, int n) {
  int i = blockIdx.x * blockDim.x + threadIdx.x;
  if (i >= n) return;
  int r = i / cols, c = i % cols;
  dst[(long)r * ld + c] = src[(long)r * ls + c];
}
__global__ void k_embed(const int* __restrict__ ph, const float* __restrict__ emb,
                        _Float16* __restrict__ dst) {
  int i = blockIdx.x * blockDim.x + threadIdx.x;
  if (i >= NTS * EPH) return;
  int tok = i / EPH, c = i % EPH;
  dst[i] = (_Float16)emb[(long)ph[tok] * EPH + c];
}
__global__ void k_spk(const int* __restrict__ spk, const float* __restrict__ emb,
                      _Float16* __restrict__ full16) {
  int i = blockIdx.x * blockDim.x + threadIdx.x;
  if (i >= NTS * ESPK) return;
  int tok = i / ESPK, c = i % ESPK, b = tok / S_;
  full16[(long)tok * FD + 2 * ENCH + c] = (_Float16)emb[(long)spk[b] * ESPK + c];
}
__global__ void k_prev(const float* __restrict__ ymels, _Float16* __restrict__ prev16) {
  int i = blockIdx.x * blockDim.x + threadIdx.x;
  if (i >= NTT * NMEL) return;
  int row = i / NMEL, c = i % NMEL, t = row % T_;
  prev16[i] = (t == 0) ? (_Float16)0.f : (_Float16)ymels[(long)(row - 1) * NMEL + c];
}
__global__ void k_final(const float* __restrict__ mel, const float* __restrict__ post,
                        float* __restrict__ out) {
  int i = blockIdx.x * blockDim.x + threadIdx.x;
  if (i < NTT * NMEL) out[i] = mel[i] + post[i];
}

// -------------------------------------------------- generic WMMA tiled GEMM
// C[M,N] = act( A[M,K] * W[N,K]^T (+ beta*C) + bias ), optional f16 mirror.
// REQUIRES: lda % 8 == 0, ldw % 8 == 0, K % 8 == 0 (caller pads).
// seqlen>0 enables the conv row-shift window on A (shift in [-2..2]).
__global__ __launch_bounds__(256)
void k_gemm(const _Float16* __restrict__ A, int lda,
            const _Float16* __restrict__ W, int ldw,
            float* __restrict__ C, int ldc,
            _Float16* __restrict__ C16, int ldc16,
            const float* __restrict__ bias,
            int M, int N, int K, int beta, int act, int seqlen, int shift) {
  __shared__ _Float16 As[64 * LDT];
  __shared__ _Float16 Bs[64 * LDT];
  int tid = threadIdx.x, lane = tid & 31, wave = tid >> 5;
  int wm = wave >> 1, wn = wave & 1;
  int bm0 = blockIdx.x * 64, bn0 = blockIdx.y * 64;
  v8f acc0 = {}, acc1 = {};

  // per-thread staging sources (loop-invariant)
  int arow = tid >> 2, acg = (tid & 3) * 8;
  int gr = bm0 + arow;
  bool arval = gr < M;
  long asrow = gr;
  if (seqlen > 0) {
    int t = gr % seqlen, ts = t + shift;
    arval = arval && (ts >= 0) && (ts < seqlen);
    asrow = (long)gr + shift;
  }
  const _Float16* Arow = A + asrow * (long)lda;
  int gn = bn0 + arow;
  bool brval = gn < N;
  const _Float16* Wrow = W + (long)gn * (long)ldw;
  _Float16* aslot = &As[arow * LDT + acg];
  _Float16* bslot = &Bs[arow * LDT + acg];

  // per-wave fragment sources (loop-invariant)
  int lr = lane & 15, ks = lane >> 4;
  const v8h* pa0 = (const v8h*)&As[(wm * 16 + lr) * LDT + ks * 16];
  const v8h* pa1 = (const v8h*)&As[(wm * 16 + lr) * LDT + ks * 16 + 8];
  const v8h* pb0l = (const v8h*)&Bs[(wn * 32 + lr) * LDT + ks * 16];
  const v8h* pb0h = (const v8h*)&Bs[(wn * 32 + lr) * LDT + ks * 16 + 8];
  const v8h* pb1l = (const v8h*)&Bs[(wn * 32 + 16 + lr) * LDT + ks * 16];
  const v8h* pb1h = (const v8h*)&Bs[(wn * 32 + 16 + lr) * LDT + ks * 16 + 8];

  for (int k0 = 0; k0 < K; k0 += 32) {
    int kk = k0 + acg;             // multiple of 8; in-bounds iff kk < K
    if (arval && kk < K) {
      async_g2l_b128(Arow + kk, aslot);          // HBM -> LDS, no VGPR bounce
      if (kk + 32 < K) __builtin_prefetch(Arow + kk + 32, 0, 0);
    } else {
      v8h z = {};
      *(v8h*)aslot = z;
    }
    if (brval && kk < K) {
      async_g2l_b128(Wrow + kk, bslot);
    } else {
      v8h z = {};
      *(v8h*)bslot = z;
    }
    async_wait_all();
    __syncthreads();
    {
      // A fragment: lanes 0-15 -> K0..15, lanes 16-31 -> K16..31 (contiguous)
      v16h af = __builtin_shufflevector(*pa0, *pa1, 0,1,2,3,4,5,6,7,8,9,10,11,12,13,14,15);
      // B fragment: lanes 0-15 -> K0..15, lanes 16-31 -> K16..31 (contiguous)
      v16h bf0 = __builtin_shufflevector(*pb0l, *pb0h, 0,1,2,3,4,5,6,7,8,9,10,11,12,13,14,15);
      v16h bf1 = __builtin_shufflevector(*pb1l, *pb1h, 0,1,2,3,4,5,6,7,8,9,10,11,12,13,14,15);
      acc0 = __builtin_amdgcn_wmma_f32_16x16x32_f16(false, af, false, bf0, (short)0, acc0, false, false);
      acc1 = __builtin_amdgcn_wmma_f32_16x16x32_f16(false, af, false, bf1, (short)0, acc1, false, false);
    }
    __syncthreads();
  }
  int row0 = bm0 + wm * 16 + ks * 8;
#pragma unroll
  for (int j = 0; j < 2; ++j) {
    v8f acc = j ? acc1 : acc0;
    int col = bn0 + wn * 32 + j * 16 + lr;
    if (col < N) {
      float bz = bias ? bias[col] : 0.f;
#pragma unroll
      for (int r = 0; r < 8; ++r) {
        int row = row0 + r;
        if (row < M) {
          float v = acc[r] + bz;
          if (beta) v += C[(long)row * ldc + col];
          if (act == ACT_RELU) v = fmaxf(v, 0.f);
          else if (act == ACT_TANH) v = tanhf(v);
          if (C)   C[(long)row * ldc + col] = v;
          if (C16) C16[(long)row * ldc16 + col] = (_Float16)v;
        }
      }
    }
  }
}

// ----------------------------------------------------------- LSTM gate stage
__global__ void k_lstm_gates(const float* __restrict__ gtmp,
                             const _Float16* __restrict__ pre,
                             int t, int seq, int H,
                             float* __restrict__ c, _Float16* __restrict__ h16,
                             _Float16* __restrict__ ys16, int ld16, int off16) {
  int idx = blockIdx.x * blockDim.x + threadIdx.x;
  if (idx >= B_ * H) return;
  int b = idx / H, j = idx % H;
  long prow = ((long)b * seq + t) * (4L * H);
  long grow = (long)b * 4 * H;
  float gi = gtmp[grow + j]         + (float)pre[prow + j];
  float gf = gtmp[grow + H + j]     + (float)pre[prow + H + j];
  float gg = gtmp[grow + 2 * H + j] + (float)pre[prow + 2 * H + j];
  float go = gtmp[grow + 3 * H + j] + (float)pre[prow + 3 * H + j];
  float cn = sigm(gf) * c[idx] + sigm(gi) * tanhf(gg);
  float h  = sigm(go) * tanhf(cn);
  c[idx] = cn;
  h16[idx] = (_Float16)h;
  ys16[((long)b * seq + t) * ld16 + off16 + j] = (_Float16)h;
}

// --------------------------------------------------------- attention pieces
__global__ void k_centers(const int* __restrict__ ydur, const float* __restrict__ ranges,
                          float* __restrict__ cpos, float* __restrict__ rinv2,
                          float* __restrict__ frames) {
  int b = blockIdx.x;
  if (threadIdx.x != 0) return;
  float cs = 0.f;
  for (int s = 0; s < S_; ++s) {
    float d = (float)ydur[b * S_ + s];
    cs += d;
    cpos[b * S_ + s] = cs - 0.5f * d;
    float r = ranges[b * S_ + s];
    rinv2[b * S_ + s] = 1.f / (r * r);
  }
  frames[b] = cs;
}
__global__ void k_att_denom(const float* __restrict__ cpos, const float* __restrict__ rinv2,
                            float* __restrict__ invden) {
  __shared__ float sc[S_], sr[S_];
  int b = blockIdx.y;
  int t = blockIdx.x * 256 + threadIdx.x;
  for (int s = threadIdx.x; s < S_; s += 256) { sc[s] = cpos[b * S_ + s]; sr[s] = rinv2[b * S_ + s]; }
  __syncthreads();
  if (t < T_) {
    float tf = (float)t, sum = 0.f;
    for (int s = 0; s < S_; ++s) {
      float d = tf - sc[s];
      sum += __expf(-sr[s] * d * d);
    }
    invden[(long)b * T_ + t] = 1.f / (sum + 1e-6f);
  }
}
// att[b,t,f] = sum_s w_norm[b,s,t] * full[b,s,f]  via WMMA (w tile built in LDS)
__global__ __launch_bounds__(256)
void k_att_apply(const float* __restrict__ cpos, const float* __restrict__ rinv2,
                 const float* __restrict__ invden, const _Float16* __restrict__ full16,
                 float* __restrict__ memf) {
  __shared__ _Float16 Wsh[64 * LDT];
  __shared__ _Float16 Fsh[64 * LDT];
  __shared__ float scp[S_], srv[S_];
  int tid = threadIdx.x, lane = tid & 31, wave = tid >> 5;
  int wm = wave >> 1, wn = wave & 1;
  int b = blockIdx.z, t0 = blockIdx.x * 64, f0 = blockIdx.y * 64;
  v8f acc0 = {}, acc1 = {};

  // preload per-batch centers / range^-2 into LDS (S_ == blockDim)
  scp[tid] = cpos[(long)b * S_ + tid];
  srv[tid] = rinv2[(long)b * S_ + tid];
  __syncthreads();

  int rr = tid >> 2, cg = (tid & 3) * 8;
  int t = t0 + rr;
  float iv = (t < T_) ? invden[(long)b * T_ + t] : 0.f;
  float tf = (float)t;
  int sl = tid >> 3, fg = (tid & 7) * 8;   // Fsh staging: s-local, f-group

  int lr = lane & 15, ks = lane >> 4;
  const v8h* pa0 = (const v8h*)&Wsh[(wm * 16 + lr) * LDT + ks * 16];
  const v8h* pa1 = (const v8h*)&Wsh[(wm * 16 + lr) * LDT + ks * 16 + 8];
  const v8h* pb0l = (const v8h*)&Fsh[(wn * 32 + lr) * LDT + ks * 16];
  const v8h* pb0h = (const v8h*)&Fsh[(wn * 32 + lr) * LDT + ks * 16 + 8];
  const v8h* pb1l = (const v8h*)&Fsh[(wn * 32 + 16 + lr) * LDT + ks * 16];
  const v8h* pb1h = (const v8h*)&Fsh[(wn * 32 + 16 + lr) * LDT + ks * 16 + 8];

  for (int s0 = 0; s0 < S_; s0 += 32) {
    // Gaussian weight tile (64 t x 32 s), normalized, f16
#pragma unroll
    for (int i = 0; i < 8; ++i) {
      int s = s0 + cg + i;
      float d = tf - scp[s];
      float wv = __expf(-srv[s] * d * d) * iv;
      Wsh[rr * LDT + cg + i] = (_Float16)wv;
    }
    // full^T tile: one 16B contiguous load along f, transposed scatter to LDS
    {
      v8h fv = *(const v8h*)&full16[((long)b * S_ + s0 + sl) * FD + f0 + fg];
#pragma unroll
      for (int i = 0; i < 8; ++i)
        Fsh[(fg + i) * LDT + sl] = fv[i];
    }
    __syncthreads();
    {
      v16h af  = __builtin_shufflevector(*pa0, *pa1, 0,1,2,3,4,5,6,7,8,9,10,11,12,13,14,15);
      v16h bf0 = __builtin_shufflevector(*pb0l, *pb0h, 0,1,2,3,4,5,6,7,8,9,10,11,12,13,14,15);
      v16h bf1 = __builtin_shufflevector(*pb1l, *pb1h, 0,1,2,3,4,5,6,7,8,9,10,11,12,13,14,15);
      acc0 = __builtin_amdgcn_wmma_f32_16x16x32_f16(false, af, false, bf0, (short)0, acc0, false, false);
      acc1 = __builtin_amdgcn_wmma_f32_16x16x32_f16(false, af, false, bf1, (short)0, acc1, false, false);
    }
    __syncthreads();
  }
  int row0 = t0 + wm * 16 + ks * 8;
#pragma unroll
  for (int j = 0; j < 2; ++j) {
    v8f acc = j ? acc1 : acc0;
    int col = f0 + wn * 32 + j * 16 + lr;
    if (col < FD) {
#pragma unroll
      for (int r = 0; r < 8; ++r) {
        int row = row0 + r;
        if (row < T_) memf[((long)b * T_ + row) * FD + col] = acc[r];
      }
    }
  }
}
// mask + positional-encoding concat -> memory f16 (B*T, 608)
__global__ void k_memfin(const float* __restrict__ memf, const float* __restrict__ frames,
                         _Float16* __restrict__ mem16) {
  int i = blockIdx.x * blockDim.x + threadIdx.x;
  if (i >= NTT * MD) return;
  int row = i / MD, col = i % MD, b = row / T_, t = row % T_;
  float v;
  if ((float)t >= frames[b]) {
    v = 0.f;
  } else if (col < FD) {
    v = memf[(long)row * FD + col];
  } else {
    int jj = col - FD, i2 = jj & ~1;
    float div = __expf(-logf(10000.f) * (float)i2 / (float)PD);
    float ang = (float)t * div;
    v = (jj & 1) ? __cosf(ang) : __sinf(ang);
  }
  mem16[i] = (_Float16)v;
}

// ==================================================================== host
extern "C" void kernel_launch(void* const* d_in, const int* in_sizes, int n_in,
                              void* d_out, int out_size, void* d_ws, size_t ws_size,
                              hipStream_t stream) {
  (void)in_sizes; (void)n_in; (void)out_size; (void)ws_size;
  auto FP = [&](int i) { return (const float*)d_in[i]; };
  auto IP = [&](int i) { return (const int*)d_in[i]; };
  // jax pytree flattening: top-level dict in insertion order for setup_inputs,
  // params dict flattened with sorted keys (lists in order, nested dicts sorted).
  const int*   phonems  = IP(0);
  const int*   speakers = IP(1);
  const int*   ydur     = IP(3);
  const float* ymels    = FP(4);
  const float* dec_bhh = FP(5), *dec_bih = FP(6), *dec_whh = FP(7), *dec_wih = FP(8);
  const float* dec_proj_b = FP(9), *dec_proj_w = FP(10);
  const float* dur_b_bhh = FP(11), *dur_b_bih = FP(12), *dur_b_whh = FP(13), *dur_b_wih = FP(14);
  const float* dur_f_bhh = FP(15), *dur_f_bih = FP(16), *dur_f_whh = FP(17), *dur_f_wih = FP(18);
  const float* dur_proj_w = FP(19);
  const float* ec_b[3] = {FP(20), FP(22), FP(24)};
  const float* ec_w[3] = {FP(21), FP(23), FP(25)};
  const float* enc_b_bhh = FP(26), *enc_b_bih = FP(27), *enc_b_whh = FP(28), *enc_b_wih = FP(29);
  const float* enc_f_bhh = FP(30), *enc_f_bih = FP(31), *enc_f_whh = FP(32), *enc_f_wih = FP(33);
  const float* phemb = FP(34);
  const float* pc_b[5] = {FP(35), FP(37), FP(39), FP(41), FP(43)};
  const float* pc_w[5] = {FP(36), FP(38), FP(40), FP(42), FP(44)};
  const float* pw1 = FP(45), *pw2 = FP(46);
  const float* rng_b_bhh = FP(47), *rng_b_bih = FP(48), *rng_b_whh = FP(49), *rng_b_wih = FP(50);
  const float* rng_f_bhh = FP(51), *rng_f_bih = FP(52), *rng_f_whh = FP(53), *rng_f_wih = FP(54);
  const float* rng_proj_b = FP(55), *rng_proj_w = FP(56);
  const float* spkemb = FP(57);

  // ---- workspace bump allocator
  char* base = (char*)d_ws;
  size_t off = 0;
  auto alloc  = [&](size_t bytes) -> void* {
    off = (off + 255) & ~(size_t)255;
    void* p = base + off; off += bytes; return p;
  };
  auto allocF = [&](size_t n) { return (float*)alloc(n * 4); };
  auto allocH = [&](size_t n) { return (_Float16*)alloc(n * 2); };

  // f16 weights
  _Float16* encw16    = allocH((size_t)3 * 5 * EPH * EPH);
  _Float16* encfwih16 = allocH((size_t)4 * ENCH * EPH);
  _Float16* encbwih16 = allocH((size_t)4 * ENCH * EPH);
  _Float16* encfwhh16 = allocH((size_t)4 * ENCH * ENCH);
  _Float16* encbwhh16 = allocH((size_t)4 * ENCH * ENCH);
  _Float16* durfwih16 = allocH((size_t)1024 * FD);
  _Float16* durbwih16 = allocH((size_t)1024 * FD);
  _Float16* durfwhh16 = allocH((size_t)1024 * 256);
  _Float16* durbwhh16 = allocH((size_t)1024 * 256);
  _Float16* rngfwih16 = allocH((size_t)1024 * RNGK);   // zero-padded K
  _Float16* rngbwih16 = allocH((size_t)1024 * RNGK);
  _Float16* rngfwhh16 = allocH((size_t)1024 * 256);
  _Float16* rngbwhh16 = allocH((size_t)1024 * 256);
  _Float16* durproj16 = allocH(512);
  _Float16* rngproj16 = allocH(512);
  _Float16* pw1_16 = allocH((size_t)PN * NMEL);
  _Float16* pw2_16 = allocH((size_t)PN * PN);
  _Float16* dwih16 = allocH((size_t)4 * DH * (PN + MD));
  _Float16* dwhh16 = allocH((size_t)4 * DH * DH);
  _Float16* dproj16 = allocH((size_t)NMEL * (DH + MD));
  _Float16* p0w16 = allocH((size_t)5 * PC * NMEL);
  _Float16* p1w16 = allocH((size_t)5 * PC * PC);
  _Float16* p2w16 = allocH((size_t)5 * PC * PC);
  _Float16* p3w16 = allocH((size_t)5 * PC * PC);
  _Float16* p4w16 = allocH((size_t)5 * NMEL * PC);
  // combined biases (bih + bhh)
  float* encfb = allocF(4 * ENCH); float* encbb = allocF(4 * ENCH);
  float* durfb = allocF(1024);     float* durbb = allocF(1024);
  float* rngfb = allocF(1024);     float* rngbb = allocF(1024);
  float* decb  = allocF(4 * DH);
  // activations
  _Float16* x16    = allocH((size_t)NTS * EPH);
  float*    convf  = allocF((size_t)NTS * EPH);
  _Float16* c16a   = allocH((size_t)NTS * EPH);
  _Float16* c16b   = allocH((size_t)NTS * EPH);
  _Float16* pre16S = allocH((size_t)NTS * 1024);
  _Float16* full16 = allocH((size_t)NTS * FD);
  _Float16* ys16   = allocH((size_t)NTS * 512);
  _Float16* rng_in16 = allocH((size_t)NTS * RNGK);     // zero-padded K
  float* rangesf = allocF(NTS);
  float* cpos    = allocF(NTS);
  float* rinv2   = allocF(NTS);
  float* frames  = allocF(B_);
  float* invden  = allocF(NTT);
  float* memf    = allocF((size_t)NTT * FD);
  _Float16* mem16   = allocH((size_t)NTT * MD);
  _Float16* prev16  = allocH((size_t)NTT * NMEL);
  _Float16* p1_16   = allocH((size_t)NTT * PN);
  _Float16* dec_in16 = allocH((size_t)NTT * (PN + MD));
  _Float16* preT16  = allocH((size_t)NTT * 4 * DH);
  _Float16* hcat16  = allocH((size_t)NTT * (DH + MD));
  float*    melf   = allocF((size_t)NTT * NMEL);
  _Float16* mel16  = allocH((size_t)NTT * NMEL);
  float*    postf  = allocF((size_t)NTT * PC);
  _Float16* post16a = allocH((size_t)NTT * PC);
  _Float16* post16b = allocH((size_t)NTT * PC);
  float*    postout = allocF((size_t)NTT * NMEL);
  float*    cvec = allocF((size_t)B_ * DH);
  _Float16* hvec = allocH((size_t)B_ * DH);
  float*    gtmp = allocF((size_t)B_ * 4 * DH);

  // ---- helpers
  auto cvt = [&](const float* src, _Float16* dst, int rows, int cols, int rs, int cs, int dld) {
    int n = rows * cols;
    k_cvt16<<<(n + 255) / 256, 256, 0, stream>>>(src, dst, n, cols, rs, cs, dld);
  };
  auto addv = [&](const float* a, const float* b, float* o, int n) {
    k_addv<<<(n + 255) / 256, 256, 0, stream>>>(a, b, o, n);
  };
  auto fillh = [&](_Float16* p, int n) {
    k_fillh<<<(n + 255) / 256, 256, 0, stream>>>(p, n);
  };
  auto gemm = [&](const _Float16* A, int lda, const _Float16* W, int ldw,
                  float* C, int ldc, _Float16* C16, int ldc16, const float* bias,
                  int M, int N, int K, int beta, int act, int seqlen, int shift) {
    dim3 g((M + 63) / 64, (N + 63) / 64);
    k_gemm<<<g, 256, 0, stream>>>(A, lda, W, ldw, C, ldc, C16, ldc16, bias,
                                  M, N, K, beta, act, seqlen, shift);
  };
  auto run_lstm_dir = [&](const _Float16* xin, int din, int seq, int Mtok,
                          const _Float16* wih16, const _Float16* whh16, const float* biasC,
                          int H, _Float16* pre16, _Float16* ysout, int ld16, int off16, bool rev) {
    gemm(xin, din, wih16, din, nullptr, 0, pre16, 4 * H, biasC, Mtok, 4 * H, din, 0, ACT_NONE, 0, 0);
    k_fillf<<<(B_ * H + 255) / 256, 256, 0, stream>>>(cvec, 0.f, B_ * H);
    fillh(hvec, B_ * H);
    for (int st = 0; st < seq; ++st) {
      int t = rev ? (seq - 1 - st) : st;
      gemm(hvec, H, whh16, H, gtmp, 4 * H, nullptr, 0, nullptr, B_, 4 * H, H, 0, ACT_NONE, 0, 0);
      k_lstm_gates<<<(B_ * H + 255) / 256, 256, 0, stream>>>(gtmp, pre16, t, seq, H,
                                                             cvec, hvec, ysout, ld16, off16);
    }
  };
  auto conv5 = [&](const _Float16* in16, int Cin, const _Float16* tapw, const float* bias,
                   float* outf, _Float16* out16, int Cout, int Mtok, int seqlen, int act) {
    for (int k = 0; k < 5; ++k) {
      bool last = (k == 4);
      gemm(in16, Cin, tapw + (size_t)k * Cout * Cin, Cin,
           outf, Cout, last ? out16 : nullptr, Cout,
           (k == 0) ? bias : nullptr, Mtok, Cout, Cin, (k == 0) ? 0 : 1,
           last ? act : ACT_NONE, seqlen, k - 2);
    }
  };

  // ---- weight conversions (f32 -> f16; conv taps gathered strided)
  for (int l = 0; l < 3; ++l)
    for (int k = 0; k < 5; ++k)
      cvt(ec_w[l] + k, encw16 + ((size_t)(l * 5 + k)) * EPH * EPH, EPH, EPH, EPH * 5, 5, EPH);
  cvt(enc_f_wih, encfwih16, 4 * ENCH, EPH, EPH, 1, EPH);
  cvt(enc_b_wih, encbwih16, 4 * ENCH, EPH, EPH, 1, EPH);
  cvt(enc_f_whh, encfwhh16, 4 * ENCH, ENCH, ENCH, 1, ENCH);
  cvt(enc_b_whh, encbwhh16, 4 * ENCH, ENCH, ENCH, 1, ENCH);
  cvt(dur_f_wih, durfwih16, 1024, FD, FD, 1, FD);
  cvt(dur_b_wih, durbwih16, 1024, FD, FD, 1, FD);
  cvt(dur_f_whh, durfwhh16, 1024, 256, 256, 1, 256);
  cvt(dur_b_whh, durbwhh16, 1024, 256, 256, 1, 256);
  // range-LSTM input weights: 1024 x 577 converted into zero-filled 1024 x 584
  fillh(rngfwih16, 1024 * RNGK);
  fillh(rngbwih16, 1024 * RNGK);
  cvt(rng_f_wih, rngfwih16, 1024, FD + 1, FD + 1, 1, RNGK);
  cvt(rng_b_wih, rngbwih16, 1024, FD + 1, FD + 1, 1, RNGK);
  cvt(rng_f_whh, rngfwhh16, 1024, 256, 256, 1, 256);
  cvt(rng_b_whh, rngbwhh16, 1024, 256, 256, 1, 256);
  cvt(dur_proj_w, durproj16, 1, 512, 512, 1, 512);
  cvt(rng_proj_w, rngproj16, 1, 512, 512, 1, 512);
  cvt(pw1, pw1_16, PN, NMEL, NMEL, 1, NMEL);
  cvt(pw2, pw2_16, PN, PN, PN, 1, PN);
  cvt(dec_wih, dwih16, 4 * DH, PN + MD, PN + MD, 1, PN + MD);
  cvt(dec_whh, dwhh16, 4 * DH, DH, DH, 1, DH);
  cvt(dec_proj_w, dproj16, NMEL, DH + MD, DH + MD, 1, DH + MD);
  for (int k = 0; k < 5; ++k) {
    cvt(pc_w[0] + k, p0w16 + (size_t)k * PC * NMEL, PC, NMEL, NMEL * 5, 5, NMEL);
    cvt(pc_w[1] + k, p1w16 + (size_t)k * PC * PC, PC, PC, PC * 5, 5, PC);
    cvt(pc_w[2] + k, p2w16 + (size_t)k * PC * PC, PC, PC, PC * 5, 5, PC);
    cvt(pc_w[3] + k, p3w16 + (size_t)k * PC * PC, PC, PC, PC * 5, 5, PC);
    cvt(pc_w[4] + k, p4w16 + (size_t)k * NMEL * PC, NMEL, PC, PC * 5, 5, PC);
  }
  addv(enc_f_bih, enc_f_bhh, encfb, 4 * ENCH);
  addv(enc_b_bih, enc_b_bhh, encbb, 4 * ENCH);
  addv(dur_f_bih, dur_f_bhh, durfb, 1024);
  addv(dur_b_bih, dur_b_bhh, durbb, 1024);
  addv(rng_f_bih, rng_f_bhh, rngfb, 1024);
  addv(rng_b_bih, rng_b_bhh, rngbb, 1024);
  addv(dec_bih, dec_bhh, decb, 4 * DH);

  // ---- 1. phoneme embedding -> x16 (B*S, 512)
  k_embed<<<(NTS * EPH + 255) / 256, 256, 0, stream>>>(phonems, phemb, x16);

  // ---- 2. encoder convs (3 layers, no activation), conv-as-5-tap-GEMM
  conv5(x16,  EPH, encw16,                          ec_b[0], convf, c16a, EPH, NTS, S_, ACT_NONE);
  conv5(c16a, EPH, encw16 + (size_t)5 * EPH * EPH,  ec_b[1], convf, c16b, EPH, NTS, S_, ACT_NONE);
  conv5(c16b, EPH, encw16 + (size_t)10 * EPH * EPH, ec_b[2], convf, c16a, EPH, NTS, S_, ACT_NONE);

  // ---- 3. encoder BiLSTM -> full16 cols [0,512)
  run_lstm_dir(c16a, EPH, S_, NTS, encfwih16, encfwhh16, encfb, ENCH, pre16S, full16, FD, 0, false);
  run_lstm_dir(c16a, EPH, S_, NTS, encbwih16, encbwhh16, encbb, ENCH, pre16S, full16, FD, ENCH, true);

  // ---- 4. speaker embedding broadcast -> full16 cols [512,576)
  k_spk<<<(NTS * ESPK + 255) / 256, 256, 0, stream>>>(speakers, spkemb, full16);

  // ---- 5. duration BiLSTM + projection (durations -> rng_in col 576)
  run_lstm_dir(full16, FD, S_, NTS, durfwih16, durfwhh16, durfb, 256, pre16S, ys16, 512, 0, false);
  run_lstm_dir(full16, FD, S_, NTS, durbwih16, durbwhh16, durbb, 256, pre16S, ys16, 512, 256, true);
  fillh(rng_in16, NTS * RNGK);   // zero pad cols [577,584)
  gemm(ys16, 512, durproj16, 512, nullptr, 0, rng_in16 + FD, RNGK, nullptr,
       NTS, 1, 512, 0, ACT_NONE, 0, 0);
  k_copy16<<<(NTS * FD + 255) / 256, 256, 0, stream>>>(full16, FD, rng_in16, RNGK, FD, NTS * FD);

  // ---- 6. range BiLSTM + projection (+bias)
  run_lstm_dir(rng_in16, RNGK, S_, NTS, rngfwih16, rngfwhh16, rngfb, 256, pre16S, ys16, 512, 0, false);
  run_lstm_dir(rng_in16, RNGK, S_, NTS, rngbwih16, rngbwhh16, rngbb, 256, pre16S, ys16, 512, 256, true);
  gemm(ys16, 512, rngproj16, 512, rangesf, 1, nullptr, 0, rng_proj_b,
       NTS, 1, 512, 0, ACT_NONE, 0, 0);

  // ---- 7. Gaussian-upsampling attention (WMMA einsum) + memory assembly
  k_centers<<<B_, 32, 0, stream>>>(ydur, rangesf, cpos, rinv2, frames);
  k_att_denom<<<dim3((T_ + 255) / 256, B_), 256, 0, stream>>>(cpos, rinv2, invden);
  k_att_apply<<<dim3((T_ + 63) / 64, FD / 64, B_), 256, 0, stream>>>(cpos, rinv2, invden, full16, memf);
  k_memfin<<<(NTT * MD + 255) / 256, 256, 0, stream>>>(memf, frames, mem16);
  k_copy16<<<(NTT * MD + 255) / 256, 256, 0, stream>>>(mem16, MD, dec_in16 + PN, PN + MD, MD, NTT * MD);
  k_copy16<<<(NTT * MD + 255) / 256, 256, 0, stream>>>(mem16, MD, hcat16 + DH, DH + MD, MD, NTT * MD);

  // ---- 8. prenet (no bias) -> dec_in cols [0,256)
  k_prev<<<(NTT * NMEL + 255) / 256, 256, 0, stream>>>(ymels, prev16);
  gemm(prev16, NMEL, pw1_16, NMEL, nullptr, 0, p1_16, PN, nullptr, NTT, PN, NMEL, 0, ACT_RELU, 0, 0);
  gemm(p1_16, PN, pw2_16, PN, nullptr, 0, dec_in16, PN + MD, nullptr, NTT, PN, PN, 0, ACT_RELU, 0, 0);

  // ---- 9. decoder LSTM: one big input projection, then 1018 recurrent steps
  gemm(dec_in16, PN + MD, dwih16, PN + MD, nullptr, 0, preT16, 4 * DH, decb,
       NTT, 4 * DH, PN + MD, 0, ACT_NONE, 0, 0);
  k_fillf<<<(B_ * DH + 255) / 256, 256, 0, stream>>>(cvec, 0.f, B_ * DH);
  fillh(hvec, B_ * DH);
  for (int t = 0; t < T_; ++t) {
    gemm(hvec, DH, dwhh16, DH, gtmp, 4 * DH, nullptr, 0, nullptr, B_, 4 * DH, DH, 0, ACT_NONE, 0, 0);
    k_lstm_gates<<<(B_ * DH + 255) / 256, 256, 0, stream>>>(gtmp, preT16, t, T_, DH,
                                                            cvec, hvec, hcat16, DH + MD, 0);
  }

  // ---- 10. mel projection
  gemm(hcat16, DH + MD, dproj16, DH + MD, melf, NMEL, mel16, NMEL, dec_proj_b,
       NTT, NMEL, DH + MD, 0, ACT_NONE, 0, 0);

  // ---- 11. postnet (4x tanh conv + final linear conv)
  conv5(mel16,   NMEL, p0w16, pc_b[0], postf, post16a, PC, NTT, T_, ACT_TANH);
  conv5(post16a, PC,   p1w16, pc_b[1], postf, post16b, PC, NTT, T_, ACT_TANH);
  conv5(post16b, PC,   p2w16, pc_b[2], postf, post16a, PC, NTT, T_, ACT_TANH);
  conv5(post16a, PC,   p3w16, pc_b[3], postf, post16b, PC, NTT, T_, ACT_TANH);
  conv5(post16b, PC,   p4w16, pc_b[4], postout, nullptr, NMEL, NTT, T_, ACT_NONE);

  // ---- 12. residual add -> output (B, T, 80) f32
  k_final<<<(NTT * NMEL + 255) / 256, 256, 0, stream>>>(melf, postout, (float*)d_out);
}
